// VisionTransformer_34136400068720
// MI455X (gfx1250) — matrix operand order repaired and checked by reading, not compiled, not run on hardware
//
#include <hip/hip_runtime.h>
#include <hip/hip_bf16.h>

typedef __attribute__((ext_vector_type(16))) _Float16 v16h;
typedef __attribute__((ext_vector_type(8)))  float    v8f;
typedef __attribute__((ext_vector_type(4)))  int      v4i;
typedef _Float16 half_t;

#if defined(__has_builtin)
#  if __has_builtin(__builtin_amdgcn_global_load_async_to_lds_b128)
#    define USE_ASYNC_LDS 1
#  endif
#endif
#ifndef USE_ASYNC_LDS
#  define USE_ASYNC_LDS 0
#endif

// typed pointers in explicit address spaces for the async-DMA builtin
typedef __attribute__((address_space(1))) v4i v4i_g;   // global
typedef __attribute__((address_space(3))) v4i v4i_l;   // LDS

__device__ __forceinline__ void wait_async(int n) {
#if USE_ASYNC_LDS
#  if defined(__has_builtin) && __has_builtin(__builtin_amdgcn_s_wait_asynccnt)
  if (n >= 8)      __builtin_amdgcn_s_wait_asynccnt(8);
  else if (n >= 4) __builtin_amdgcn_s_wait_asynccnt(4);
  else             __builtin_amdgcn_s_wait_asynccnt(0);
#  else
  if (n >= 8)      asm volatile("s_wait_asynccnt 8" ::: "memory");
  else if (n >= 4) asm volatile("s_wait_asynccnt 4" ::: "memory");
  else             asm volatile("s_wait_asynccnt 0" ::: "memory");
#  endif
#else
  (void)n;
#endif
}

#define D_MODEL 768
#define NHEAD   12
#define DHEAD   64
#define NLAYER  12
#define FFDIM   3072
#define NPATCH  576
#define BATCH   16
#define NTOK    (BATCH * NPATCH)   // 9216
#define KPATCH  1200
#define KPAD    1216               // padded to multiple of 32
#define WP_     18                 // 360/20
#define PS      20

// ---------------------------------------------------------------------------
// WMMA GEMM: C[m,n] = sum_k A[m,k] * W[n,k]  (+bias[n]) (+resid) (relu)
//   BM x BN block tile, WR x WC wave grid (32*WR*WC threads), K-step 32.
//   Each wave computes (BM/WR/16) x (BN/WC/16) v_wmma_f32_16x16x32_f16 tiles.
// Staging: 4-stage LDS ring, 2 stages in flight via CDNA5 async global->LDS
// DMA (ASYNCcnt), one barrier per K-step; fallback stages through registers.
// Requires M%BM==0, N%BN==0, K%32==0 (arranged by the launcher).
// ---------------------------------------------------------------------------
struct GemmP {
  const half_t* A;        // [M,K]  lda
  const half_t* B;        // [N,K]  ldb   (weight-style, row-major along K)
  const float*  bias;     // [N] or null
  const float*  resid;    // [M,ldc] f32 or null
  float*        Cf;       // f32 out or null
  half_t*       Ch;       // f16 out or null
  int M, N, K, lda, ldb, ldc;
  long sAo, sAi, sBo, sBi, sCo, sCi;   // batch strides (elements)
  int  zInner;                          // z = zo*zInner + zi
};

// OUTMODE: 0 = plain (ldc layout), 1 = heads [B,H,N,DH], 2 = headsT [B,H,DH,N]
template<int BM, int BN, int WR, int WC, int OUTMODE, bool RELU>
__global__ void __launch_bounds__(32 * WR * WC, 1) gemm_wmma(GemmP p) {
  constexpr int T    = 32 * WR * WC;
  constexpr int WMT  = BM / (WR * 16);     // WMMA M-tiles per wave
  constexpr int WNT  = BN / (WC * 16);     // WMMA N-tiles per wave
  constexpr int AST  = 40;                 // padded LDS row stride (halves)
  constexpr int NCHA = (BM * 4) / T;       // uint4 chunks per thread (A)
  constexpr int NCHB = (BN * 4) / T;       // uint4 chunks per thread (B)
  constexpr int NSTG = 4;                  // LDS ring stages (2 in flight)
  constexpr int NASY = NCHA + NCHB;        // async ops per stage per thread

  __shared__ __align__(16) half_t As[NSTG][BM * AST];
  __shared__ __align__(16) half_t Bs[NSTG][BN * AST];

  const int tid  = threadIdx.x;
  const int lane = tid & 31;
  const int wid  = tid >> 5;
  const int wm   = wid / WC;
  const int wn   = wid % WC;
  const int bm   = blockIdx.y * BM;
  const int bn   = blockIdx.x * BN;

  const int z  = blockIdx.z;
  const int zo = z / p.zInner;
  const int zi = z % p.zInner;
  const half_t* __restrict__ A  = p.A + zo * p.sAo + zi * p.sAi;
  const half_t* __restrict__ Bw = p.B + zo * p.sBo + zi * p.sBi;
  const long    coff = zo * p.sCo + zi * p.sCi;

  const int r  = lane & 15;
  const int hi = lane >> 4;

  v8f acc[WMT][WNT] = {};

#if USE_ASYNC_LDS
  auto stage = [&](int k, int stg) {
#pragma unroll
    for (int i = 0; i < NCHA; ++i) {
      int c = tid + i * T, row = c >> 2, sub = c & 3;
      __builtin_amdgcn_global_load_async_to_lds_b128(
          (v4i_g*)(A + (long)(bm + row) * p.lda + k + sub * 8),
          (v4i_l*)&As[stg][row * AST + sub * 8], 0, 0);
    }
#pragma unroll
    for (int i = 0; i < NCHB; ++i) {
      int c = tid + i * T, row = c >> 2, sub = c & 3;
      __builtin_amdgcn_global_load_async_to_lds_b128(
          (v4i_g*)(Bw + (long)(bn + row) * p.ldb + k + sub * 8),
          (v4i_l*)&Bs[stg][row * AST + sub * 8], 0, 0);
    }
  };
#else
  uint4 ra[NCHA], rb[NCHB];
  auto gload = [&](int k) {
#pragma unroll
    for (int i = 0; i < NCHA; ++i) {
      int c = tid + i * T, row = c >> 2, sub = c & 3;
      ra[i] = *(const uint4*)(A + (long)(bm + row) * p.lda + k + sub * 8);
    }
#pragma unroll
    for (int i = 0; i < NCHB; ++i) {
      int c = tid + i * T, row = c >> 2, sub = c & 3;
      rb[i] = *(const uint4*)(Bw + (long)(bn + row) * p.ldb + k + sub * 8);
    }
  };
  auto sstore = [&](int stg) {
#pragma unroll
    for (int i = 0; i < NCHA; ++i) {
      int c = tid + i * T, row = c >> 2, sub = c & 3;
      *(uint4*)&As[stg][row * AST + sub * 8] = ra[i];
    }
#pragma unroll
    for (int i = 0; i < NCHB; ++i) {
      int c = tid + i * T, row = c >> 2, sub = c & 3;
      *(uint4*)&Bs[stg][row * AST + sub * 8] = rb[i];
    }
  };
#endif

  auto compute = [&](int stg) {
    union Frag { v16h v; uint4 q[2]; };
    Frag a[WMT], b[WNT];
#pragma unroll
    for (int mi = 0; mi < WMT; ++mi) {
      int arow = wm * WMT * 16 + mi * 16 + r;
      // lane<16: K {0..7,16..23}; lane>=16: K {8..15,24..31}
      a[mi].q[0] = *(const uint4*)&As[stg][arow * AST + hi * 8];
      a[mi].q[1] = *(const uint4*)&As[stg][arow * AST + 16 + hi * 8];
    }
#pragma unroll
    for (int ni = 0; ni < WNT; ++ni) {
      int bcol = wn * WNT * 16 + ni * 16 + r;
      // lane<16: K {0..7,16..23}; lane>=16: K {8..15,24..31}
      b[ni].q[0] = *(const uint4*)&Bs[stg][bcol * AST + hi * 8];
      b[ni].q[1] = *(const uint4*)&Bs[stg][bcol * AST + 16 + hi * 8];
    }
#pragma unroll
    for (int mi = 0; mi < WMT; ++mi)
#pragma unroll
      for (int ni = 0; ni < WNT; ++ni)
        acc[mi][ni] = __builtin_amdgcn_wmma_f32_16x16x32_f16(
            false, a[mi].v, false, b[ni].v, (short)0, acc[mi][ni],
            false, false);
  };

  const int kiters = p.K / 32;
#if USE_ASYNC_LDS
  stage(0, 0);
  if (kiters > 1) stage(32, 1);
  for (int j = 0; j < kiters; ++j) {
    if (j + 2 < kiters) stage((j + 2) * 32, (j + 2) % NSTG);  // depth-2 DMA
    int pend = kiters - 1 - j; if (pend > 2) pend = 2;        // stages ahead
    wait_async(pend * NASY);                                  // stage j landed
    __syncthreads();
    compute(j % NSTG);
  }
#else
  gload(0);
  sstore(0);
  __syncthreads();
  for (int j = 0; j < kiters; ++j) {
    const bool more = (j + 1) < kiters;
    if (more) {
      gload((j + 1) * 32);                           // in flight behind WMMAs
      if ((j + 2) < kiters) {                        // gfx1250 global_prefetch_b8
        __builtin_prefetch(A + (long)(bm + (tid % BM)) * p.lda + (j + 2) * 32, 0, 1);
        __builtin_prefetch(Bw + (long)(bn + (tid % BN)) * p.ldb + (j + 2) * 32, 0, 1);
      }
    }
    compute(j % NSTG);
    if (more) {
      sstore((j + 1) % NSTG);
      __syncthreads();
    }
  }
#endif

  // ---- epilogue: VGPR v holds M = v + 8*(lane>=16), N = lane&15 ----
#pragma unroll
  for (int mi = 0; mi < WMT; ++mi)
#pragma unroll
    for (int ni = 0; ni < WNT; ++ni)
#pragma unroll
      for (int v = 0; v < 8; ++v) {
        int row = bm + wm * WMT * 16 + mi * 16 + hi * 8 + v;
        int col = bn + wn * WNT * 16 + ni * 16 + r;
        float val = acc[mi][ni][v];
        if (p.bias)  val += p.bias[col];
        if (p.resid) val += p.resid[(long)row * p.ldc + col];
        if (RELU)    val = val > 0.f ? val : 0.f;
        if (OUTMODE == 0) {
          long idx = coff + (long)row * p.ldc + col;
          if (p.Cf) p.Cf[idx] = val;
          if (p.Ch) p.Ch[idx] = (half_t)val;
        } else {
          int bb = row / NPATCH, tok = row % NPATCH;
          int hh = col >> 6, dd = col & 63;
          long idx;
          if (OUTMODE == 1)        // [B,H,N,DH]
            idx = (((long)(bb * NHEAD + hh) * NPATCH + tok) << 6) + dd;
          else                     // [B,H,DH,N]
            idx = ((long)(bb * NHEAD + hh) * DHEAD + dd) * NPATCH + tok;
          p.Ch[idx] = (half_t)val;
        }
      }
}

// ---------------------------------------------------------------------------
// im2col: x[16,3,640,360] -> A0 f16 [9216, 1216] (K padded with zeros)
// ---------------------------------------------------------------------------
__global__ void im2col_k(const float* __restrict__ x, half_t* __restrict__ out) {
  long idx = (long)blockIdx.x * blockDim.x + threadIdx.x;
  if (idx >= (long)NTOK * KPAD) return;
  int m = (int)(idx / KPAD), kk = (int)(idx % KPAD);
  float v = 0.f;
  if (kk < KPATCH) {
    int b = m / NPATCH, n = m % NPATCH;
    int hp = n / WP_, wp = n % WP_;
    int c = kk / (PS * PS), rem = kk % (PS * PS);
    int py = rem / PS, px = rem % PS;
    v = x[(((long)b * 3 + c) * 640 + hp * PS + py) * 360 + wp * PS + px];
  }
  out[idx] = (half_t)v;
}

// f32 -> f16 with optional zero-padding of the inner dim
__global__ void cvt_pad_k(half_t* __restrict__ dst, const float* __restrict__ src,
                          int rows, int scols, int dcols) {
  long idx = (long)blockIdx.x * blockDim.x + threadIdx.x;
  if (idx >= (long)rows * dcols) return;
  int rr = (int)(idx / dcols), cc = (int)(idx % dcols);
  dst[idx] = (half_t)(cc < scols ? src[(long)rr * scols + cc] : 0.f);
}

// LayerNorm over D of (h + pos_emb); writes f32 residual stream + f16 copy
__global__ void __launch_bounds__(256) ln_k(float* __restrict__ h,
                                            const float* __restrict__ pos,
                                            const float* __restrict__ g,
                                            const float* __restrict__ bta,
                                            half_t* __restrict__ h16) {
  __shared__ float buf[D_MODEL];
  __shared__ float red[256];
  int row = blockIdx.x, tid = threadIdx.x;
  int tok = row % NPATCH;
  float s = 0.f;
  for (int i = tid; i < D_MODEL; i += 256) {
    float v = h[(long)row * D_MODEL + i] + pos[(long)tok * D_MODEL + i];
    buf[i] = v; s += v;
  }
  red[tid] = s; __syncthreads();
  for (int o = 128; o > 0; o >>= 1) { if (tid < o) red[tid] += red[tid + o]; __syncthreads(); }
  float mu = red[0] / D_MODEL; __syncthreads();
  s = 0.f;
  for (int i = tid; i < D_MODEL; i += 256) { float d = buf[i] - mu; s += d * d; }
  red[tid] = s; __syncthreads();
  for (int o = 128; o > 0; o >>= 1) { if (tid < o) red[tid] += red[tid + o]; __syncthreads(); }
  float inv = rsqrtf(red[0] / D_MODEL + 1e-5f);
  for (int i = tid; i < D_MODEL; i += 256) {
    float o = (buf[i] - mu) * inv * g[i] + bta[i];
    h[(long)row * D_MODEL + i]   = o;
    h16[(long)row * D_MODEL + i] = (half_t)o;
  }
}

// row softmax of scores*0.125 -> f16 probs; one block per row of 576
__global__ void __launch_bounds__(256) softmax_k(const float* __restrict__ sc,
                                                 half_t* __restrict__ pr) {
  __shared__ float buf[NPATCH];
  __shared__ float red[256];
  long base = (long)blockIdx.x * NPATCH;
  int tid = threadIdx.x;
  float m = -1e30f;
  for (int i = tid; i < NPATCH; i += 256) { float v = sc[base + i] * 0.125f; buf[i] = v; m = fmaxf(m, v); }
  red[tid] = m; __syncthreads();
  for (int o = 128; o > 0; o >>= 1) { if (tid < o) red[tid] = fmaxf(red[tid], red[tid + o]); __syncthreads(); }
  m = red[0]; __syncthreads();
  float s = 0.f;
  for (int i = tid; i < NPATCH; i += 256) { float e = __expf(buf[i] - m); buf[i] = e; s += e; }
  red[tid] = s; __syncthreads();
  for (int o = 128; o > 0; o >>= 1) { if (tid < o) red[tid] += red[tid + o]; __syncthreads(); }
  float inv = 1.f / red[0];
  for (int i = tid; i < NPATCH; i += 256) pr[base + i] = (half_t)(buf[i] * inv);
}

// mean over tokens + 10-class FC; one block per batch image
__global__ void __launch_bounds__(256) poolfc_k(const float* __restrict__ h,
                                                const float* __restrict__ fw,
                                                const float* __restrict__ fb,
                                                float* __restrict__ out) {
  __shared__ float pooled[D_MODEL];
  int b = blockIdx.x, tid = threadIdx.x;
  for (int d = tid; d < D_MODEL; d += 256) {
    float s = 0.f;
    for (int t = 0; t < NPATCH; ++t) s += h[(long)(b * NPATCH + t) * D_MODEL + d];
    pooled[d] = s * (1.f / NPATCH);
  }
  __syncthreads();
  if (tid < 10) {
    float a = fb[tid];
    for (int d = 0; d < D_MODEL; ++d) a += pooled[d] * fw[tid * D_MODEL + d];
    out[b * 10 + tid] = a;
  }
}

// ---------------------------------------------------------------------------
extern "C" void kernel_launch(void* const* d_in, const int* in_sizes, int n_in,
                              void* d_out, int out_size, void* d_ws, size_t ws_size,
                              hipStream_t stream) {
  (void)in_sizes; (void)n_in; (void)out_size; (void)ws_size;
  const float* x       = (const float*)d_in[0];
  const float* patch_w = (const float*)d_in[1];
  const float* patch_b = (const float*)d_in[2];
  const float* pos_emb = (const float*)d_in[3];
  const float* ln_g    = (const float*)d_in[4];
  const float* ln_b    = (const float*)d_in[5];
  const float* Wq = (const float*)d_in[6],  *bq = (const float*)d_in[7];
  const float* Wk = (const float*)d_in[8],  *bk = (const float*)d_in[9];
  const float* Wv = (const float*)d_in[10], *bv = (const float*)d_in[11];
  const float* Wo = (const float*)d_in[12], *bo = (const float*)d_in[13];
  const float* W1 = (const float*)d_in[14], *b1 = (const float*)d_in[15];
  const float* W2 = (const float*)d_in[16], *b2 = (const float*)d_in[17];
  const float* fc_w = (const float*)d_in[18], *fc_b = (const float*)d_in[19];

  char* wsp = (char*)d_ws;
  size_t off = 0;
  auto alloc = [&](size_t bytes) -> char* {
    char* p = wsp + off; off += (bytes + 255) & ~(size_t)255; return p;
  };
  float*  hF    = (float*) alloc((size_t)NTOK * D_MODEL * 4);
  half_t* h16   = (half_t*)alloc((size_t)NTOK * D_MODEL * 2);
  half_t* a0    = (half_t*)alloc((size_t)NTOK * KPAD * 2);
  half_t* wbuf  = (half_t*)alloc((size_t)FFDIM * D_MODEL * 2);
  half_t* q16   = (half_t*)alloc((size_t)NTOK * D_MODEL * 2);
  half_t* k16   = (half_t*)alloc((size_t)NTOK * D_MODEL * 2);
  half_t* v16   = (half_t*)alloc((size_t)NTOK * D_MODEL * 2);
  float*  scor  = (float*) alloc((size_t)BATCH * NHEAD * NPATCH * NPATCH * 4);
  half_t* probs = (half_t*)alloc((size_t)BATCH * NHEAD * NPATCH * NPATCH * 2);
  half_t* ffh   = (half_t*)alloc((size_t)NTOK * FFDIM * 2);
  half_t* o16   = (half_t*)alloc((size_t)NTOK * D_MODEL * 2);

  auto cvt = [&](half_t* dst, const float* src, int rows, int scols, int dcols) {
    long n = (long)rows * dcols;
    cvt_pad_k<<<dim3((unsigned)((n + 255) / 256)), 256, 0, stream>>>(dst, src, rows, scols, dcols);
  };
  auto mkP = [&](const half_t* A, const half_t* B, const float* bias, const float* resid,
                 float* Cf, half_t* Ch, int M, int N, int K, int lda, int ldb, int ldc) {
    GemmP p{};
    p.A = A; p.B = B; p.bias = bias; p.resid = resid; p.Cf = Cf; p.Ch = Ch;
    p.M = M; p.N = N; p.K = K; p.lda = lda; p.ldb = ldb; p.ldc = ldc;
    p.sAo = p.sAi = p.sBo = p.sBi = p.sCo = p.sCi = 0; p.zInner = 1;
    return p;
  };

  // ---- patch embed ----
  {
    long n = (long)NTOK * KPAD;
    im2col_k<<<dim3((unsigned)((n + 255) / 256)), 256, 0, stream>>>(x, a0);
  }
  cvt(wbuf, patch_w, D_MODEL, KPATCH, KPAD);
  {
    GemmP p = mkP(a0, wbuf, patch_b, nullptr, hF, nullptr,
                  NTOK, D_MODEL, KPAD, KPAD, KPAD, D_MODEL);
    gemm_wmma<128, 128, 2, 4, 0, false>
        <<<dim3(D_MODEL / 128, NTOK / 128, 1), 256, 0, stream>>>(p);
  }
  ln_k<<<NTOK, 256, 0, stream>>>(hF, pos_emb, ln_g, ln_b, h16);

  // ---- transformer layers ----
  for (int l = 0; l < NLAYER; ++l) {
    const size_t wsz = (size_t)D_MODEL * D_MODEL;
    // Q, K, V projections (head-split outputs; V transposed per head)
    cvt(wbuf, Wq + (size_t)l * wsz, D_MODEL, D_MODEL, D_MODEL);
    {
      GemmP p = mkP(h16, wbuf, bq + (size_t)l * D_MODEL, nullptr, nullptr, q16,
                    NTOK, D_MODEL, D_MODEL, D_MODEL, D_MODEL, D_MODEL);
      gemm_wmma<128, 128, 2, 4, 1, false>
          <<<dim3(D_MODEL / 128, NTOK / 128, 1), 256, 0, stream>>>(p);
    }
    cvt(wbuf, Wk + (size_t)l * wsz, D_MODEL, D_MODEL, D_MODEL);
    {
      GemmP p = mkP(h16, wbuf, bk + (size_t)l * D_MODEL, nullptr, nullptr, k16,
                    NTOK, D_MODEL, D_MODEL, D_MODEL, D_MODEL, D_MODEL);
      gemm_wmma<128, 128, 2, 4, 1, false>
          <<<dim3(D_MODEL / 128, NTOK / 128, 1), 256, 0, stream>>>(p);
    }
    cvt(wbuf, Wv + (size_t)l * wsz, D_MODEL, D_MODEL, D_MODEL);
    {
      GemmP p = mkP(h16, wbuf, bv + (size_t)l * D_MODEL, nullptr, nullptr, v16,
                    NTOK, D_MODEL, D_MODEL, D_MODEL, D_MODEL, D_MODEL);
      gemm_wmma<128, 128, 2, 4, 2, false>
          <<<dim3(D_MODEL / 128, NTOK / 128, 1), 256, 0, stream>>>(p);
    }
    // scores = Q @ K^T, batched over 192 (b,h)
    {
      GemmP p = mkP(q16, k16, nullptr, nullptr, scor, nullptr,
                    NPATCH, NPATCH, DHEAD, DHEAD, DHEAD, NPATCH);
      p.sAo = (long)NPATCH * DHEAD;
      p.sBo = (long)NPATCH * DHEAD;
      p.sCo = (long)NPATCH * NPATCH;
      gemm_wmma<64, 64, 2, 2, 0, false>
          <<<dim3(NPATCH / 64, NPATCH / 64, BATCH * NHEAD), 128, 0, stream>>>(p);
    }
    softmax_k<<<BATCH * NHEAD * NPATCH, 256, 0, stream>>>(scor, probs);
    // O = P @ V  (V stored [B,H,DH,N] so both operands are K-contiguous)
    {
      GemmP p = mkP(probs, v16, nullptr, nullptr, nullptr, o16,
                    NPATCH, DHEAD, NPATCH, NPATCH, NPATCH, D_MODEL);
      p.zInner = NHEAD;
      p.sAo = (long)NHEAD * NPATCH * NPATCH; p.sAi = (long)NPATCH * NPATCH;
      p.sBo = (long)NHEAD * DHEAD * NPATCH;  p.sBi = (long)DHEAD * NPATCH;
      p.sCo = (long)NPATCH * D_MODEL;        p.sCi = DHEAD;
      gemm_wmma<64, 64, 2, 2, 0, false>
          <<<dim3(1, NPATCH / 64, BATCH * NHEAD), 128, 0, stream>>>(p);
    }
    // output projection + residual -> updates h (f32) and h16
    cvt(wbuf, Wo + (size_t)l * wsz, D_MODEL, D_MODEL, D_MODEL);
    {
      GemmP p = mkP(o16, wbuf, bo + (size_t)l * D_MODEL, hF, hF, h16,
                    NTOK, D_MODEL, D_MODEL, D_MODEL, D_MODEL, D_MODEL);
      gemm_wmma<128, 128, 2, 4, 0, false>
          <<<dim3(D_MODEL / 128, NTOK / 128, 1), 256, 0, stream>>>(p);
    }
    // FFN1: relu(h @ W1^T + b1) -> f16
    cvt(wbuf, W1 + (size_t)l * FFDIM * D_MODEL, FFDIM, D_MODEL, D_MODEL);
    {
      GemmP p = mkP(h16, wbuf, b1 + (size_t)l * FFDIM, nullptr, nullptr, ffh,
                    NTOK, FFDIM, D_MODEL, D_MODEL, D_MODEL, FFDIM);
      gemm_wmma<128, 128, 2, 4, 0, true>
          <<<dim3(FFDIM / 128, NTOK / 128, 1), 256, 0, stream>>>(p);
    }
    // FFN2 + residual -> h (f32) and h16
    cvt(wbuf, W2 + (size_t)l * D_MODEL * FFDIM, D_MODEL, FFDIM, FFDIM);
    {
      GemmP p = mkP(ffh, wbuf, b2 + (size_t)l * D_MODEL, hF, hF, h16,
                    NTOK, D_MODEL, FFDIM, FFDIM, FFDIM, D_MODEL);
      gemm_wmma<128, 128, 2, 4, 0, false>
          <<<dim3(D_MODEL / 128, NTOK / 128, 1), 256, 0, stream>>>(p);
    }
  }

  // ---- mean pool + classifier ----
  poolfc_k<<<BATCH, 256, 0, stream>>>(hF, fc_w, fc_b, (float*)d_out);
}
